// GraphMambaGMN_35313221107838
// MI455X (gfx1250) — matrix operand order repaired and checked by reading, not compiled, not run on hardware
//
#include <hip/hip_runtime.h>
#include <hip/hip_bf16.h>

// ---------------------------------------------------------------------------
// GraphMamba GMN on MI455X (gfx1250).
// All dense GEMMs go through v_wmma_f32_16x16x32_f16 (f32 accumulate).
// GEMM is software-pipelined: double-buffered LDS, tile k+1 global loads
// issued before the fragment reads + WMMA of tile k, one barrier/iteration.
// Hot instantiations (K%32==0) have branch-free unconditional b128 loads.
// ---------------------------------------------------------------------------

typedef __attribute__((ext_vector_type(16))) _Float16 v16h;
typedef __attribute__((ext_vector_type(8)))  float    v8f;

namespace gmn {
constexpr int B_    = 4;
constexpr int N_    = 512;
constexpr int CIN   = 128;
constexpr int DM    = 256;   // d_model
constexpr int DI    = 512;   // d_inner
constexpr int DS    = 16;    // d_state
constexpr int DTR   = 16;    // dt_rank
constexpr int BN    = B_ * N_;   // 2048 tokens
constexpr int DBC   = DTR + 2 * DS;  // 48
}

__device__ __forceinline__ float silu_f(float v) {
  return v / (1.0f + __expf(-v));
}

// ---------------------------------------------------------------------------
// Generic WMMA GEMM:  C[b] = alpha*act(A[b] @ W[b] + bias) + beta*C[b]
//   A: [M,K] row major (lda), or transposed access A[k*lda+m] if TRANSA
//   W: [K,N] row major
// Preconditions: M%16==0, N%16==0, K%4==0, lda%4==0, N%4==0, 16B alignment.
// FULLK: K%32==0 (no masking anywhere).
// 128 threads = 4 waves; block covers 16(M) x 64(N); wave = one 16x16 tile.
// act: 0=none, 1=relu, 2=softplus
// ---------------------------------------------------------------------------
template <bool TRANSA, bool FULLK>
__global__ __launch_bounds__(128)
void wmma_gemm_kernel(const float* __restrict__ A, const float* __restrict__ W,
                      const float* __restrict__ bias, float* __restrict__ C,
                      int M, int N, int K, int lda,
                      int strideA, int strideB, int strideC,
                      float alpha, float beta, int act)
{
  const int tid  = threadIdx.x;
  const int lane = tid & 31;
  const int wave = tid >> 5;
  const int m0 = blockIdx.y * 16;
  const int n0 = (blockIdx.x * 4 + wave) * 16;

  const float* Ab = A + (long long)blockIdx.z * strideA;
  const float* Wb = W + (long long)blockIdx.z * strideB;
  float*       Cb = C + (long long)blockIdx.z * strideC;

  __shared__ _Float16 As[2][16][32];        // [buf][m][k]
  __shared__ _Float16 Bs[2][4][16][32];     // [buf][wave][n][k] (transposed)

  v8f acc = {};

  // load-pattern coordinates
  const int a_r  = tid >> 3;          // !TRANSA: row, float4 along k
  const int a_c4 = (tid & 7) * 4;
  const int a_kk = tid >> 2;          // TRANSA: k, float4 along m
  const int a_m4 = (tid & 3) * 4;
  const int b_ng = (lane & 3) * 4;    // B: n-group (same for all 4 loads)
  const int b_k0 = lane >> 2;         // B: k = b_k0 + 8*i
  // clamp the B column group so the edge wave of a non-multiple-of-64 N
  // (x_proj: N=48) never reads past the weight allocation; its stores are
  // guarded in the epilogue, so the duplicate data is harmless.
  const int b_gn = min(n0 + b_ng, N - 4);

  // running pointers (incremented per K-tile; no per-iteration 64-bit muls)
  const float* pa_cur = TRANSA ? (Ab + (long long)a_kk * lda + (m0 + a_m4))
                               : (Ab + (long long)(m0 + a_r) * lda + a_c4);
  const long long stepA = TRANSA ? (long long)32 * lda : 32;
  const float* pb_cur = Wb + (long long)b_k0 * N + b_gn;
  const long long stepB = (long long)32 * N;
  const long long bOff  = (long long)8 * N;

  const int nk = (K + 31) / 32;

  float4 av;
  float4 bv[4];

  auto load_tile = [&](int kt) {
    if (FULLK) {
      av = *(const float4*)pa_cur;
#pragma unroll
      for (int i = 0; i < 4; ++i)
        bv[i] = *(const float4*)(pb_cur + i * bOff);
      pa_cur += stepA;
      pb_cur += stepB;
    } else {
      int k0 = kt * 32;
      if (!TRANSA) {
        int gk  = k0 + a_c4;
        int gkc = min(gk, K - 4);
        av = *(const float4*)(Ab + (long long)(m0 + a_r) * lda + gkc);
        if (gk + 0 >= K) av.x = 0.0f;
        if (gk + 1 >= K) av.y = 0.0f;
        if (gk + 2 >= K) av.z = 0.0f;
        if (gk + 3 >= K) av.w = 0.0f;
      } else {
        int gk  = k0 + a_kk;
        int gkc = min(gk, K - 1);
        av = *(const float4*)(Ab + (long long)gkc * lda + (m0 + a_m4));
        if (gk >= K) { av.x = 0.0f; av.y = 0.0f; av.z = 0.0f; av.w = 0.0f; }
      }
#pragma unroll
      for (int i = 0; i < 4; ++i) {
        int gk  = k0 + b_k0 + 8 * i;
        int gkc = min(gk, K - 1);
        bv[i] = *(const float4*)(Wb + (long long)gkc * N + b_gn);
        if (gk >= K) { bv[i].x = 0.0f; bv[i].y = 0.0f; bv[i].z = 0.0f; bv[i].w = 0.0f; }
      }
    }
  };

  auto store_tile = [&](int p) {
    if (!TRANSA) {
      union { _Float16 h[4]; unsigned long long u; } pa;
      pa.h[0] = (_Float16)av.x; pa.h[1] = (_Float16)av.y;
      pa.h[2] = (_Float16)av.z; pa.h[3] = (_Float16)av.w;
      *(unsigned long long*)&As[p][a_r][a_c4] = pa.u;     // ds_store_b64
    } else {
      As[p][a_m4 + 0][a_kk] = (_Float16)av.x;
      As[p][a_m4 + 1][a_kk] = (_Float16)av.y;
      As[p][a_m4 + 2][a_kk] = (_Float16)av.z;
      As[p][a_m4 + 3][a_kk] = (_Float16)av.w;
    }
#pragma unroll
    for (int i = 0; i < 4; ++i) {
      int kk = b_k0 + 8 * i;
      Bs[p][wave][b_ng + 0][kk] = (_Float16)bv[i].x;
      Bs[p][wave][b_ng + 1][kk] = (_Float16)bv[i].y;
      Bs[p][wave][b_ng + 2][kk] = (_Float16)bv[i].z;
      Bs[p][wave][b_ng + 3][kk] = (_Float16)bv[i].w;
    }
  };

  // ---- prologue: tile 0 ----
  load_tile(0);
  store_tile(0);
  __syncthreads();

  const int mrow   = lane & 15;
  const int kbaseA = (lane < 16) ? 0 : 16;   // A: lanes0-15 K{0-15}
  const int kbaseB = (lane < 16) ? 0 : 16;   // B: lanes0-15 K{0-15}

  for (int kt = 0; kt < nk; ++kt) {
    const int p = kt & 1;
    if (kt + 1 < nk) load_tile(kt + 1);      // overlap with LDS reads + WMMA

    // ---- build fragments (contiguous k = 16*(lane/16) + s, same for A and B) ----
    v16h af, bfr;
#pragma unroll
    for (int v = 0; v < 8; ++v) {
      int kA = kbaseA + 2 * v;
      af[2 * v]     = As[p][mrow][kA];
      af[2 * v + 1] = As[p][mrow][kA + 1];
      int kB = kbaseB + 2 * v;
      bfr[2 * v]     = Bs[p][wave][mrow][kB];
      bfr[2 * v + 1] = Bs[p][wave][mrow][kB + 1];
    }

    acc = __builtin_amdgcn_wmma_f32_16x16x32_f16(
        false, af, false, bfr, (short)0, acc, false, false);

    if (kt + 1 < nk) {
      store_tile(p ^ 1);    // waits on this iteration's global loads here
      __syncthreads();
    }
  }

  // --- epilogue: C/D layout lanes0-15 rows 0-7, lanes16-31 rows 8-15 ---
  const int n = lane & 15;
  const int gn = n0 + n;
  const int mbase = (lane < 16) ? 0 : 8;
  if (gn < N) {
    float bv2 = bias ? bias[gn] : 0.0f;
#pragma unroll
    for (int r = 0; r < 8; ++r) {
      int gm = m0 + mbase + r;
      float v = acc[r] + bv2;
      if (act == 1)      v = fmaxf(v, 0.0f);
      else if (act == 2) v = (v > 20.0f) ? v : log1pf(__expf(v));
      long long o = (long long)gm * N + gn;
      float prev = (beta != 0.0f) ? beta * Cb[o] : 0.0f;
      Cb[o] = alpha * v + prev;
    }
  }
}

// ---------------------------------------------------------------------------
// LayerNorm over last dim (D=256), one 256-thread block per row.
// ---------------------------------------------------------------------------
__global__ __launch_bounds__(256)
void layernorm_kernel(const float* __restrict__ x, const float* __restrict__ w,
                      const float* __restrict__ b, float* __restrict__ out, int D)
{
  const int row = blockIdx.x;
  const int t = threadIdx.x;
  __shared__ float red[256];
  float v = x[(long long)row * D + t];
  red[t] = v;
  __syncthreads();
  for (int s = 128; s > 0; s >>= 1) { if (t < s) red[t] += red[t + s]; __syncthreads(); }
  float mu = red[0] / (float)D;
  __syncthreads();
  float dv = v - mu;
  red[t] = dv * dv;
  __syncthreads();
  for (int s = 128; s > 0; s >>= 1) { if (t < s) red[t] += red[t + s]; __syncthreads(); }
  float var = red[0] / (float)D;
  out[(long long)row * D + t] = dv * rsqrtf(var + 1e-5f) * w[t] + b[t];
}

// ---------------------------------------------------------------------------
// Causal depthwise conv (K=4) + SiLU, with direction folding:
// for dir=1 we compute conv on the time-reversed sequence and write results
// back at original positions (equivalent to flip->conv->...->flip).
// xi = xz[:, :di] with row stride xzstride.
// ---------------------------------------------------------------------------
__global__ __launch_bounds__(256)
void conv_silu_kernel(const float* __restrict__ xz, const float* __restrict__ w,
                      const float* __restrict__ bias, float* __restrict__ xc,
                      int Bseq, int L, int di, int xzstride, int dir)
{
  long long idx = (long long)blockIdx.x * blockDim.x + threadIdx.x;
  long long total = (long long)Bseq * L * di;
  if (idx >= total) return;
  int c = (int)(idx % di);
  long long tl = idx / di;
  int l = (int)(tl % L);
  int s = (int)(tl / L);
  int v = dir ? (L - 1 - l) : l;          // virtual time of this output
  float acc = 0.0f;
#pragma unroll
  for (int k = 0; k < 4; ++k) {
    int vv = v - 3 + k;
    if (vv >= 0) {
      int srcl = dir ? (L - 1 - vv) : vv;
      acc += xz[((long long)s * L + srcl) * xzstride + c] * w[c * 4 + k];
    }
  }
  acc += bias[c];
  xc[idx] = acc * (1.0f / (1.0f + __expf(-acc)));
}

// ---------------------------------------------------------------------------
// Selective scan. One thread per (seq, channel); 16-state vector in registers.
// dbc row layout: [dt_raw(16) | B(16) | C(16)].  y = sum_s h*C + xc*D.
// dir folds the bidirectional flip into the time loop.
// ---------------------------------------------------------------------------
__global__ __launch_bounds__(256)
void scan_kernel(const float* __restrict__ dt, const float* __restrict__ xc,
                 const float* __restrict__ dbc, const float* __restrict__ A_log,
                 const float* __restrict__ Dp, float* __restrict__ y,
                 int Bseq, int L, int di, int dir)
{
  long long idx = (long long)blockIdx.x * blockDim.x + threadIdx.x;
  long long total = (long long)Bseq * di;
  if (idx >= total) return;
  int c = (int)(idx % di);
  int s = (int)(idx / di);

  float Arow[16];
#pragma unroll
  for (int j = 0; j < 16; ++j) Arow[j] = -__expf(A_log[c * 16 + j]);
  float Dv = Dp[c];
  float h[16];
#pragma unroll
  for (int j = 0; j < 16; ++j) h[j] = 0.0f;

  for (int v = 0; v < L; ++v) {
    int l = dir ? (L - 1 - v) : v;
    long long t = (long long)s * L + l;
    float dtv = dt[t * di + c];
    float xcv = xc[t * di + c];
    const float4* rB = (const float4*)(dbc + t * gmn::DBC + 16);
    const float4* rC = (const float4*)(dbc + t * gmn::DBC + 32);
    float Bv[16], Cv[16];
#pragma unroll
    for (int q = 0; q < 4; ++q) {
      float4 tb = rB[q], tc = rC[q];
      Bv[4 * q + 0] = tb.x; Bv[4 * q + 1] = tb.y; Bv[4 * q + 2] = tb.z; Bv[4 * q + 3] = tb.w;
      Cv[4 * q + 0] = tc.x; Cv[4 * q + 1] = tc.y; Cv[4 * q + 2] = tc.z; Cv[4 * q + 3] = tc.w;
    }
    // prefetch next timestep (global_prefetch_b8)
    if (v + 1 < L) {
      int l2 = dir ? (L - 2 - v) : (v + 1);
      long long t2 = (long long)s * L + l2;
      __builtin_prefetch(dt + t2 * di + c, 0, 0);
      __builtin_prefetch(dbc + t2 * gmn::DBC, 0, 0);
    }
    float accy = 0.0f;
#pragma unroll
    for (int j = 0; j < 16; ++j) {
      float dA = __expf(dtv * Arow[j]);
      h[j] = h[j] * dA + dtv * Bv[j] * xcv;
      accy += h[j] * Cv[j];
    }
    y[t * di + c] = accy + xcv * Dv;
  }
}

// ---------------------------------------------------------------------------
// Gate: y <- y * silu(z), z = xz[:, di + c]
// ---------------------------------------------------------------------------
__global__ __launch_bounds__(256)
void gate_kernel(float* __restrict__ y, const float* __restrict__ xz,
                 long long total, int di, int xzstride)
{
  long long idx = (long long)blockIdx.x * blockDim.x + threadIdx.x;
  if (idx >= total) return;
  long long t = idx / di;
  int c = (int)(idx % di);
  float z = xz[t * xzstride + di + c];
  y[idx] = y[idx] * silu_f(z);
}

// ---------------------------------------------------------------------------
// Concat: cat[r] = [x[r] (c1) | m[r] (c2)]
// ---------------------------------------------------------------------------
__global__ __launch_bounds__(256)
void concat_kernel(const float* __restrict__ x, const float* __restrict__ m,
                   float* __restrict__ cat, int rows, int c1, int c2)
{
  long long idx = (long long)blockIdx.x * blockDim.x + threadIdx.x;
  long long total = (long long)rows * (c1 + c2);
  if (idx >= total) return;
  int ct = c1 + c2;
  int r = (int)(idx / ct);
  int c = (int)(idx % ct);
  cat[idx] = (c < c1) ? x[(long long)r * c1 + c] : m[(long long)r * c2 + (c - c1)];
}

// ---------------------------------------------------------------------------
// Host side
// ---------------------------------------------------------------------------
namespace {

struct StackParams {
  const float *ln_w, *ln_b, *in_proj, *conv_w, *conv_b, *x_proj,
              *dt_w, *dt_b, *A_log, *D, *out_proj;
};

inline void gemm(hipStream_t st, const float* A, const float* W, const float* bias,
                 float* C, int M, int N, int K, int lda, int transA,
                 int sA, int sB, int sC, int batch,
                 float alpha, float beta, int act)
{
  dim3 grid((N + 63) / 64, (M + 15) / 16, batch);
  if (transA)
    wmma_gemm_kernel<true, true><<<grid, 128, 0, st>>>(
        A, W, bias, C, M, N, K, lda, sA, sB, sC, alpha, beta, act);
  else if ((K & 31) == 0)
    wmma_gemm_kernel<false, true><<<grid, 128, 0, st>>>(
        A, W, bias, C, M, N, K, lda, sA, sB, sC, alpha, beta, act);
  else
    wmma_gemm_kernel<false, false><<<grid, 128, 0, st>>>(
        A, W, bias, C, M, N, K, lda, sA, sB, sC, alpha, beta, act);
}

inline dim3 pw_grid(long long total) { return dim3((unsigned)((total + 255) / 256)); }

} // namespace

extern "C" void kernel_launch(void* const* d_in, const int* in_sizes, int n_in,
                              void* d_out, int out_size, void* d_ws, size_t ws_size,
                              hipStream_t stream)
{
  using namespace gmn;
  const float* x      = (const float*)d_in[0];   // [4,512,128]
  const float* adj    = (const float*)d_in[1];   // [4,512,512]
  const float* msg_w1 = (const float*)d_in[3];
  const float* msg_b1 = (const float*)d_in[4];
  const float* msg_w2 = (const float*)d_in[5];
  const float* msg_b2 = (const float*)d_in[6];
  const float* upd_w1 = (const float*)d_in[7];
  const float* upd_b1 = (const float*)d_in[8];
  const float* upd_w2 = (const float*)d_in[9];
  const float* upd_b2 = (const float*)d_in[10];

  auto load_stack = [&](int base) {
    StackParams p;
    p.ln_w   = (const float*)d_in[base + 0];
    p.ln_b   = (const float*)d_in[base + 1];
    p.in_proj= (const float*)d_in[base + 2];
    p.conv_w = (const float*)d_in[base + 3];
    p.conv_b = (const float*)d_in[base + 4];
    p.x_proj = (const float*)d_in[base + 5];
    p.dt_w   = (const float*)d_in[base + 6];
    p.dt_b   = (const float*)d_in[base + 7];
    p.A_log  = (const float*)d_in[base + 8];
    p.D      = (const float*)d_in[base + 9];
    p.out_proj=(const float*)d_in[base + 10];
    return p;
  };
  StackParams loc = load_stack(11);
  StackParams glb = load_stack(22);

  // workspace layout (floats)
  float* ws    = (float*)d_ws;
  float* bufA  = ws;                       // [BN, DM]
  float* bufB  = bufA + (long long)BN * DM;
  float* catb  = bufB + (long long)BN * DM;      // [BN, CIN+DM]
  float* hbuf  = catb + (long long)BN * (CIN + DM);
  float* lobuf = hbuf + (long long)BN * DM;
  float* xnb   = lobuf + (long long)BN * DM;
  float* xzb   = xnb  + (long long)BN * DM;      // [BN, 2*DI]
  float* xcb   = xzb  + (long long)BN * 2 * DI;  // [BN, DI]
  float* dbcb  = xcb  + (long long)BN * DI;      // [BN, 48]
  float* dtb   = dbcb + (long long)BN * DBC;     // [BN, DI]
  float* ybuf  = dtb  + (long long)BN * DI;      // [BN, DI]

  // ---------------- MPNN ----------------
  // H1 = relu(x @ msg_w1 + b1)
  gemm(stream, x, msg_w1, msg_b1, bufA, BN, DM, CIN, CIN, 0, 0, 0, 0, 1, 1.0f, 0.0f, 1);
  // msg = H1 @ msg_w2 + b2
  gemm(stream, bufA, msg_w2, msg_b2, bufB, BN, DM, DM, DM, 0, 0, 0, 0, 1, 1.0f, 0.0f, 0);
  // m[b] = adj[b]^T @ msg[b]   (batched, A transposed)
  gemm(stream, adj, bufB, nullptr, bufA, N_, DM, N_, N_, 1,
       N_ * N_, N_ * DM, N_ * DM, B_, 1.0f, 0.0f, 0);
  // cat = [x | m]
  concat_kernel<<<pw_grid((long long)BN * (CIN + DM)), 256, 0, stream>>>(
      x, bufA, catb, BN, CIN, DM);
  // H2 = relu(cat @ upd_w1 + b1)
  gemm(stream, catb, upd_w1, upd_b1, bufB, BN, DM, CIN + DM, CIN + DM, 0, 0, 0, 0, 1, 1.0f, 0.0f, 1);
  // neigh = H2 @ upd_w2 + b2
  gemm(stream, bufB, upd_w2, upd_b2, hbuf, BN, DM, DM, DM, 0, 0, 0, 0, 1, 1.0f, 0.0f, 0);

  // ---------------- Bidirectional Mamba stacks ----------------
  float* hcur = hbuf;
  float* lout = lobuf;

  auto run_stack = [&](const StackParams& p, int nlayers, int Bseq, int L) {
    for (int i = 0; i < nlayers; ++i) {
      layernorm_kernel<<<BN, 256, 0, stream>>>(hcur, p.ln_w + i * DM, p.ln_b + i * DM, xnb, DM);
      for (int dd = 0; dd < 2; ++dd) {
        int pi = i * 2 + dd;
        // xz = xn @ in_proj    [BN, 2*DI]
        gemm(stream, xnb, p.in_proj + (long long)pi * DM * 2 * DI, nullptr, xzb,
             BN, 2 * DI, DM, DM, 0, 0, 0, 0, 1, 1.0f, 0.0f, 0);
        // xc = silu(causal_conv(xi) + conv_b)
        conv_silu_kernel<<<pw_grid((long long)BN * DI), 256, 0, stream>>>(
            xzb, p.conv_w + (long long)pi * DI * 4, p.conv_b + (long long)pi * DI,
            xcb, Bseq, L, DI, 2 * DI, dd);
        // dbc = xc @ x_proj    [BN, 48]
        gemm(stream, xcb, p.x_proj + (long long)pi * DI * DBC, nullptr, dbcb,
             BN, DBC, DI, DI, 0, 0, 0, 0, 1, 1.0f, 0.0f, 0);
        // dt = softplus(dbc[:, :16] @ dt_w + dt_b)   [BN, DI]
        gemm(stream, dbcb, p.dt_w + (long long)pi * DTR * DI, p.dt_b + (long long)pi * DI,
             dtb, BN, DI, DTR, DBC, 0, 0, 0, 0, 1, 1.0f, 0.0f, 2);
        // selective scan -> y (includes + xc*D)
        scan_kernel<<<pw_grid((long long)Bseq * DI), 256, 0, stream>>>(
            dtb, xcb, dbcb, p.A_log + (long long)pi * DI * DS,
            p.D + (long long)pi * DI, ybuf, Bseq, L, DI, dd);
        // y *= silu(z)
        gate_kernel<<<pw_grid((long long)BN * DI), 256, 0, stream>>>(
            ybuf, xzb, (long long)BN * DI, DI, 2 * DI);
        // out += 0.5 * (y @ out_proj)
        gemm(stream, ybuf, p.out_proj + (long long)pi * DI * DM, nullptr, lout,
             BN, DM, DI, DI, 0, 0, 0, 0, 1, 0.5f, dd ? 1.0f : 0.0f, 0);
      }
      float* t = hcur; hcur = lout; lout = t;
    }
  };

  run_stack(loc, 2, BN, 1);     // local: 2048 sequences of length 1
  run_stack(glb, 2, B_, N_);    // global: 4 sequences of length 512

  hipMemcpyAsync(d_out, hcur, (long long)BN * DM * sizeof(float),
                 hipMemcpyDeviceToDevice, stream);
}
